// Attention_84086869721556
// MI455X (gfx1250) — matrix operand order repaired and checked by reading, not compiled, not run on hardware
//
#include <hip/hip_runtime.h>
#include <hip/hip_bf16.h>
#include <math.h>

// ---------------------------------------------------------------------------
// GQA attention (B=1, S=2048, DIM=4096, NH=32, NKV=8, HD=128) for gfx1250.
// All matmul-shaped work via v_wmma_f32_16x16x32_bf16 (wave32).
// Attention computes the score tile TRANSPOSED (K as WMMA-A, Q as WMMA-B) so
// the softmax'd P tile lands directly in WMMA A-fragment layout in registers:
// no LDS round trip, and row reductions are a single shfl_xor(16).
// ---------------------------------------------------------------------------

#define S_LEN 2048
#define DIM   4096
#define NH    32
#define NKV   8
#define HD    128
#define NOUT  6144          // (NH + 2*NKV) * HD

typedef __attribute__((ext_vector_type(16))) __bf16 v16bf;
typedef __attribute__((ext_vector_type(8)))  float  v8f;

union Frag {
    v16bf bf;
    unsigned short us[16];
    uint4 q[2];
};

__device__ __forceinline__ unsigned short f32_to_bf16(float f) {
    unsigned int u = __builtin_bit_cast(unsigned int, f);
    unsigned int r = u + 0x7FFFu + ((u >> 16) & 1u);   // round-to-nearest-even
    return (unsigned short)(r >> 16);
}
__device__ __forceinline__ float bf16_to_f32(unsigned short h) {
    unsigned int u = ((unsigned int)h) << 16;
    return __builtin_bit_cast(float, u);
}

// ---------------------------------------------------------------------------
// Kernel 1: fp32 -> bf16 convert
// ---------------------------------------------------------------------------
__global__ void cvt_f32_to_bf16(const float* __restrict__ in,
                                unsigned short* __restrict__ out, int n) {
    int i = blockIdx.x * blockDim.x + threadIdx.x;
    if (i < n) out[i] = f32_to_bf16(in[i]);
}

// ---------------------------------------------------------------------------
// Kernels 2/5: GEMM  C[M][N] = A[M][K] @ W[N][K]^T.
// Wave computes 32(M) x 64(N): 2 A-fragments shared across 4 B-fragments,
// 8 WMMA per K-step of 32 (1.5 b128 loads per WMMA).
// Block = 4 waves => 32 x 256 tile.
// ---------------------------------------------------------------------------
template <int N_TOTAL, bool BF16_OUT>
__device__ __forceinline__
void gemm_body(const unsigned short* __restrict__ A,
               const unsigned short* __restrict__ W,
               void* __restrict__ Cv)
{
    const int K = DIM;
    const int wave = threadIdx.x >> 5;
    const int lane = threadIdx.x & 31;
    const int mrow = lane & 15;
    const int half = lane >> 4;
    const int m0 = blockIdx.x * 32;
    const int n0 = blockIdx.y * 256 + wave * 64;

    const unsigned short* arow0 = A + (size_t)(m0 + mrow) * K + half * 8;
    const unsigned short* arow1 = arow0 + (size_t)16 * K;
    const unsigned short* brow[4];
#pragma unroll
    for (int t = 0; t < 4; ++t)
        brow[t] = W + (size_t)(n0 + 16 * t + mrow) * K + half * 8;

    v8f z = {0.f,0.f,0.f,0.f,0.f,0.f,0.f,0.f};
    v8f acc0[4], acc1[4];
#pragma unroll
    for (int t = 0; t < 4; ++t) { acc0[t] = z; acc1[t] = z; }

    for (int k = 0; k < K; k += 32) {
        Frag a0, a1;
        a0.q[0] = *(const uint4*)(arow0 + k);
        a0.q[1] = *(const uint4*)(arow0 + k + 16);
        a1.q[0] = *(const uint4*)(arow1 + k);
        a1.q[1] = *(const uint4*)(arow1 + k + 16);
#pragma unroll
        for (int t = 0; t < 4; ++t) {
            Frag b;
            b.q[0] = *(const uint4*)(brow[t] + k);
            b.q[1] = *(const uint4*)(brow[t] + k + 16);
            acc0[t] = __builtin_amdgcn_wmma_f32_16x16x32_bf16(
                false, a0.bf, false, b.bf, (short)0, acc0[t], false, false);
            acc1[t] = __builtin_amdgcn_wmma_f32_16x16x32_bf16(
                false, a1.bf, false, b.bf, (short)0, acc1[t], false, false);
        }
    }
#pragma unroll
    for (int t = 0; t < 4; ++t)
#pragma unroll
        for (int j = 0; j < 8; ++j) {
            int m = 8 * half + j;
            size_t i0 = (size_t)(m0 + m) * N_TOTAL + n0 + 16 * t + mrow;
            size_t i1 = (size_t)(m0 + 16 + m) * N_TOTAL + n0 + 16 * t + mrow;
            if (BF16_OUT) {
                ((unsigned short*)Cv)[i0] = f32_to_bf16(acc0[t][j]);
                ((unsigned short*)Cv)[i1] = f32_to_bf16(acc1[t][j]);
            } else {
                ((float*)Cv)[i0] = acc0[t][j];
                ((float*)Cv)[i1] = acc1[t][j];
            }
        }
}

__global__ __launch_bounds__(128)
void gemm_qkv(const unsigned short* __restrict__ A,
              const unsigned short* __restrict__ W,
              unsigned short* __restrict__ C) {
    gemm_body<NOUT, true>(A, W, C);
}

__global__ __launch_bounds__(128)
void gemm_out(const unsigned short* __restrict__ A,
              const unsigned short* __restrict__ W,
              float* __restrict__ C) {
    gemm_body<DIM, false>(A, W, C);
}

// ---------------------------------------------------------------------------
// Kernel 3: RoPE for Q and K heads only (V handled by v_transpose).
//   Q[h][s][hd] (scaled by 1/sqrt(HD)), K[g][s][hd]     (bf16)
// ---------------------------------------------------------------------------
__global__ void rope_qk(const unsigned short* __restrict__ qkv, // [S][NOUT]
                        const float* __restrict__ fc,           // [S][64][2]
                        unsigned short* __restrict__ Qb,
                        unsigned short* __restrict__ Kb)
{
    int idx = blockIdx.x * blockDim.x + threadIdx.x;
    if (idx >= S_LEN * 40 * 64) return;
    int pair = idx & 63;
    int hh   = (idx >> 6) % 40;
    int s    = idx / (40 * 64);

    const unsigned short* src = qkv + (size_t)s * NOUT + hh * HD + pair * 2;
    float x0 = bf16_to_f32(src[0]), x1 = bf16_to_f32(src[1]);
    float c  = fc[(s * 64 + pair) * 2 + 0];
    float sn = fc[(s * 64 + pair) * 2 + 1];
    float re = x0 * c - x1 * sn;
    float im = x0 * sn + x1 * c;
    if (hh < NH) {
        const float scale = 0.088388347648318447f; // 1/sqrt(128)
        unsigned short* dst = Qb + ((size_t)hh * S_LEN + s) * HD + pair * 2;
        dst[0] = f32_to_bf16(re * scale);
        dst[1] = f32_to_bf16(im * scale);
    } else {
        int g = hh - NH;
        unsigned short* dst = Kb + ((size_t)g * S_LEN + s) * HD + pair * 2;
        dst[0] = f32_to_bf16(re);
        dst[1] = f32_to_bf16(im);
    }
}

// ---------------------------------------------------------------------------
// Kernel 4: coalesced V transpose via LDS tile: qkv V region -> Vt[g][hd][s]
// 64x64 bf16 tile, pad to 66 (stride 33 banks, gcd(33,64)=1 -> conflict-free).
// ---------------------------------------------------------------------------
__global__ __launch_bounds__(256)
void v_transpose(const unsigned short* __restrict__ qkv,
                 unsigned short* __restrict__ Vtb)
{
    __shared__ unsigned short tile[64][66];
    const int tx = threadIdx.x;       // 0..63
    const int ty = threadIdx.y;       // 0..3
    const int s0 = blockIdx.x * 64;
    const int d0 = blockIdx.y * 64;
    const int g  = blockIdx.z;
    const int vbase = (NH + NKV) * HD; // 5120

#pragma unroll
    for (int i = 0; i < 16; ++i) {
        int sl = i * 4 + ty;
        tile[sl][tx] = qkv[(size_t)(s0 + sl) * NOUT + vbase + g * HD + d0 + tx];
    }
    __syncthreads();
#pragma unroll
    for (int i = 0; i < 16; ++i) {
        int dl = i * 4 + ty;
        Vtb[((size_t)g * HD + d0 + dl) * S_LEN + s0 + tx] = tile[tx][dl];
    }
}

// ---------------------------------------------------------------------------
// Kernel 5: causal flash attention, one wave per (head, 16-query tile).
// Scores computed TRANSPOSED: S^T[key][q] = K_tile(A) x Q^T(B).
// C'-layout: lane holds query q = lane&15; its 16 score values are keys
// {8*half+0..7} U {16+8*half+0..7} == the WMMA A-fragment pattern for P.
// So P is built in registers, softmax row stats need one shfl_xor(16).
// ---------------------------------------------------------------------------
__global__ __launch_bounds__(128)
void attn_fwd(const unsigned short* __restrict__ Qb,   // [NH][S][HD]
              const unsigned short* __restrict__ Kb,   // [NKV][S][HD]
              const unsigned short* __restrict__ Vtb,  // [NKV][HD][S]
              unsigned short* __restrict__ O)          // [S][NH*HD]
{
    const int wave = threadIdx.x >> 5;
    const int lane = threadIdx.x & 31;
    const int mrow = lane & 15;
    const int half = lane >> 4;

    const int q0 = blockIdx.x * 16;
    const int h  = blockIdx.y * 4 + wave;   // 4 heads per block, same kv group
    const int g  = h >> 2;                  // rep = NH/NKV = 4

    // Q as B-fragment: lane = query q0+mrow, elems = hd (kk + half*8 + 0..7, kk + 16 + half*8 + 0..7).
    // Loop-invariant over key tiles: hoist all 4 K-step fragments.
    const unsigned short* qrow = Qb + ((size_t)h * S_LEN + q0 + mrow) * HD + half * 8;
    Frag qf[4];
#pragma unroll
    for (int kk = 0; kk < 4; ++kk) {
        qf[kk].q[0] = *(const uint4*)(qrow + kk * 32);
        qf[kk].q[1] = *(const uint4*)(qrow + kk * 32 + 16);
    }

    v8f z = {0.f,0.f,0.f,0.f,0.f,0.f,0.f,0.f};
    v8f acc[8];
#pragma unroll
    for (int t = 0; t < 8; ++t) acc[t] = z;
    float mstat = -3.0e38f, lstat = 0.f;

    const int q_g  = q0 + mrow;      // this lane's query (for masking/stats)
    const int kend = q0 + 15;

    for (int j0 = 0; j0 <= kend; j0 += 32) {
        // ---- S^T tiles: c0 = keys [j0,j0+16), c1 = keys [j0+16,j0+32) ----
        v8f c0 = z, c1 = z;
        const unsigned short* krow0 =
            Kb + ((size_t)g * S_LEN + j0 + mrow) * HD + half * 8;
        const unsigned short* krow1 = krow0 + (size_t)16 * HD;
#pragma unroll
        for (int kk = 0; kk < 4; ++kk) {
            Frag a0, a1;
            a0.q[0] = *(const uint4*)(krow0 + kk * 32);
            a0.q[1] = *(const uint4*)(krow0 + kk * 32 + 16);
            a1.q[0] = *(const uint4*)(krow1 + kk * 32);
            a1.q[1] = *(const uint4*)(krow1 + kk * 32 + 16);
            c0 = __builtin_amdgcn_wmma_f32_16x16x32_bf16(
                false, a0.bf, false, qf[kk].bf, (short)0, c0, false, false);
            c1 = __builtin_amdgcn_wmma_f32_16x16x32_bf16(
                false, a1.bf, false, qf[kk].bf, (short)0, c1, false, false);
        }

        // ---- causal mask + per-lane (per-query) online softmax ----
        float v0[8], v1[8];
        float mx = -3.0e38f;
#pragma unroll
        for (int j = 0; j < 8; ++j) {
            int k0g = j0 + 8 * half + j;        // key of c0[j]
            int k1g = k0g + 16;                 // key of c1[j]
            v0[j] = (k0g <= q_g) ? c0[j] : -3.0e38f;
            v1[j] = (k1g <= q_g) ? c1[j] : -3.0e38f;
            mx = fmaxf(mx, fmaxf(v0[j], v1[j]));
        }
        mx = fmaxf(mx, __shfl_xor(mx, 16, 32));       // combine lane ^ 16
        float mn    = fmaxf(mstat, mx);
        float alpha = __expf(mstat - mn);
        mstat = mn;

        Frag p;                                        // P in A-frag layout
        float ls = 0.f;
#pragma unroll
        for (int j = 0; j < 8; ++j) {
            float p0 = __expf(v0[j] - mn);
            float p1 = __expf(v1[j] - mn);
            ls += p0 + p1;
            p.us[j]     = f32_to_bf16(p0);
            p.us[8 + j] = f32_to_bf16(p1);
        }
        ls += __shfl_xor(ls, 16, 32);
        lstat = alpha * lstat + ls;

        // redistribute alpha to accumulator row mapping (row m = 8*half+j)
        float aj[8];
#pragma unroll
        for (int j = 0; j < 8; ++j) aj[j] = __shfl(alpha, 8 * half + j, 32);
#pragma unroll
        for (int t = 0; t < 8; ++t)
#pragma unroll
            for (int j = 0; j < 8; ++j) acc[t][j] = acc[t][j] * aj[j];

        // ---- acc += P @ V_tile (V as B-fragment from Vt[g][hd][s]) ----
#pragma unroll
        for (int t = 0; t < 8; ++t) {
            Frag b;
            const unsigned short* vrow =
                Vtb + ((size_t)g * HD + 16 * t + mrow) * S_LEN + j0 + half * 8;
            b.q[0] = *(const uint4*)(vrow);
            b.q[1] = *(const uint4*)(vrow + 16);
            acc[t] = __builtin_amdgcn_wmma_f32_16x16x32_bf16(
                false, p.bf, false, b.bf, (short)0, acc[t], false, false);
        }
    }

    // ---- epilogue: normalize rows, store bf16 attn output [s][h*HD + d] ----
    float linv = 1.0f / lstat;
    float rj[8];
#pragma unroll
    for (int j = 0; j < 8; ++j) rj[j] = __shfl(linv, 8 * half + j, 32);
#pragma unroll
    for (int j = 0; j < 8; ++j) {
        int m_g = q0 + 8 * half + j;
#pragma unroll
        for (int t = 0; t < 8; ++t) {
            O[(size_t)m_g * (NH * HD) + h * HD + 16 * t + mrow] =
                f32_to_bf16(acc[t][j] * rj[j]);
        }
    }
}

// ---------------------------------------------------------------------------
// Launch
// ---------------------------------------------------------------------------
extern "C" void kernel_launch(void* const* d_in, const int* in_sizes, int n_in,
                              void* d_out, int out_size, void* d_ws, size_t ws_size,
                              hipStream_t stream) {
    const float* x    = (const float*)d_in[0];   // [S][DIM]
    const float* fc   = (const float*)d_in[1];   // [S][64][2]
    const float* wqkv = (const float*)d_in[2];   // [NOUT][DIM]
    const float* wo   = (const float*)d_in[3];   // [DIM][DIM]
    float* out = (float*)d_out;

    // workspace carve (bytes), lifetime-based aliasing:
    //   xb:   [0,        16MB)  x bf16        (dead after gemm_qkv -> Ob)
    //   wqb:  [16MB,     64MB)  wqkv bf16     (dead after gemm_qkv -> wob)
    //   qkvb: [64MB,     88MB)  qkv bf16      (dead after rope/v_transpose)
    //   Qb:   [88MB,    104MB)
    //   Kb:   [104MB,   108MB)
    //   Vtb:  [108MB,   112MB)
    char* ws = (char*)d_ws;
    unsigned short* xb   = (unsigned short*)(ws);
    unsigned short* wqb  = (unsigned short*)(ws + (size_t)16777216);
    unsigned short* qkvb = (unsigned short*)(ws + (size_t)67108864);
    unsigned short* Qb   = (unsigned short*)(ws + (size_t)92274688);
    unsigned short* Kb   = (unsigned short*)(ws + (size_t)109051904);
    unsigned short* Vtb  = (unsigned short*)(ws + (size_t)113246208);
    unsigned short* wob  = wqb;  // reuse after gemm_qkv
    unsigned short* Ob   = xb;   // reuse after gemm_qkv

    const int nX  = S_LEN * DIM;    // 8388608
    const int nWq = NOUT * DIM;     // 25165824
    const int nWo = DIM * DIM;      // 16777216

    cvt_f32_to_bf16<<<(nX  + 255) / 256, 256, 0, stream>>>(x,    xb,  nX);
    cvt_f32_to_bf16<<<(nWq + 255) / 256, 256, 0, stream>>>(wqkv, wqb, nWq);

    gemm_qkv<<<dim3(S_LEN / 32, NOUT / 256), 128, 0, stream>>>(xb, wqb, qkvb);

    cvt_f32_to_bf16<<<(nWo + 255) / 256, 256, 0, stream>>>(wo, wob, nWo);

    rope_qk<<<(S_LEN * 40 * 64) / 256, 256, 0, stream>>>(qkvb, fc, Qb, Kb);
    v_transpose<<<dim3(S_LEN / 64, HD / 64, NKV), dim3(64, 4), 0, stream>>>(qkvb, Vtb);

    attn_fwd<<<dim3(S_LEN / 16, NH / 4), 128, 0, stream>>>(Qb, Kb, Vtb, Ob);

    gemm_out<<<dim3(S_LEN / 32, DIM / 256), 128, 0, stream>>>(Ob, wob, out);
}
